// BagKQMClassModel_2439541424386
// MI455X (gfx1250) — matrix operand order repaired and checked by reading, not compiled, not run on hardware
//
#include <hip/hip_runtime.h>

typedef __attribute__((ext_vector_type(16))) _Float16 v16h;
typedef __attribute__((ext_vector_type(8)))  float    v8f;

#define BS_  256
#define N_   128
#define DX_  32
#define DY_  10
#define M_   2048

// One block per batch element. 256 threads = 8 waves (wave32).
// Each wave: keeps all 8 A row-tiles (16x32 f16) resident, loops over its 16
// B column-tiles of c_x, does v_wmma_f32_16x16x32_f16 per (row-tile, col-tile),
// applies the RBF epilogue and reduces into 11 LDS accumulators.
__global__ __launch_bounds__(256) void kqm_fused_kernel(
    const float* __restrict__ inputs,   // (BS, N, DX)
    const float* __restrict__ sigma_p,  // scalar
    const float* __restrict__ c_x,      // (M, DX)
    const float* __restrict__ c_y,      // (M, DY)
    const float* __restrict__ comp_w,   // (M,)
    float* __restrict__ out)            // (BS, DY)
{
    __shared__ float s_a2[N_];
    __shared__ float s_red[DY_ + 1];

    const int b    = blockIdx.x;
    const int tid  = threadIdx.x;
    const int lane = tid & 31;
    const int wave = tid >> 5;       // 0..7
    const int hi   = lane >> 4;      // lane half
    const int lr   = lane & 15;

    const float* binp = inputs + (size_t)b * (N_ * DX_);

    // Stage ||a||^2 for this batch's 128 rows, zero reduction slots.
    if (tid < N_) {
        const float* row = binp + tid * DX_;
        float s = 0.f;
#pragma unroll
        for (int k = 0; k < DX_; ++k) { float x = row[k]; s += x * x; }
        s_a2[tid] = s;
    }
    if (tid < DY_ + 1) s_red[tid] = 0.f;
    __syncthreads();

    const float sg  = fmaxf(sigma_p[0], 1e-3f);
    const float e2s = 1.4426950408889634f / (sg * sg);  // log2(e)/sigma^2

    // ---- Load 8 A fragments (ISA 16-bit A 16x32 layout: lane half selects
    // K base 0 or 8; each half holds K {kb..kb+7, kb+16..kb+23}).
    const int akb = hi ? 8 : 0;
    v16h afrag[8];
#pragma unroll
    for (int it = 0; it < 8; ++it) {
        const float* ar = binp + (it * 16 + lr) * DX_ + akb;
        float4 f0 = *(const float4*)(ar + 0);
        float4 f1 = *(const float4*)(ar + 4);
        float4 f2 = *(const float4*)(ar + 16);
        float4 f3 = *(const float4*)(ar + 20);
        v16h a;
        a[0]  = (_Float16)f0.x; a[1]  = (_Float16)f0.y;
        a[2]  = (_Float16)f0.z; a[3]  = (_Float16)f0.w;
        a[4]  = (_Float16)f1.x; a[5]  = (_Float16)f1.y;
        a[6]  = (_Float16)f1.z; a[7]  = (_Float16)f1.w;
        a[8]  = (_Float16)f2.x; a[9]  = (_Float16)f2.y;
        a[10] = (_Float16)f2.z; a[11] = (_Float16)f2.w;
        a[12] = (_Float16)f3.x; a[13] = (_Float16)f3.y;
        a[14] = (_Float16)f3.z; a[15] = (_Float16)f3.w;
        afrag[it] = a;
    }

    float t_loc = 0.f;
    float p_loc[DY_];
#pragma unroll
    for (int d = 0; d < DY_; ++d) p_loc[d] = 0.f;

    // ---- Loop over this wave's m-tiles (16 of 128).
    for (int mt = wave; mt < M_ / 16; mt += 8) {
        const int m = mt * 16 + lr;   // this lane's output column / c_x row

        // B fragment: column m of B == row m of c_x; lane half holds
        // K {kb..kb+7, kb+16..kb+23}, kb = 8*hi (same order as A).
        const float* br = c_x + (size_t)m * DX_ + hi * 8;
        float4 g0 = *(const float4*)(br + 0);
        float4 g1 = *(const float4*)(br + 4);
        float4 g2 = *(const float4*)(br + 16);
        float4 g3 = *(const float4*)(br + 20);
        v16h bv;
        bv[0]  = (_Float16)g0.x; bv[1]  = (_Float16)g0.y;
        bv[2]  = (_Float16)g0.z; bv[3]  = (_Float16)g0.w;
        bv[4]  = (_Float16)g1.x; bv[5]  = (_Float16)g1.y;
        bv[6]  = (_Float16)g1.z; bv[7]  = (_Float16)g1.w;
        bv[8]  = (_Float16)g2.x; bv[9]  = (_Float16)g2.y;
        bv[10] = (_Float16)g2.z; bv[11] = (_Float16)g2.w;
        bv[12] = (_Float16)g3.x; bv[13] = (_Float16)g3.y;
        bv[14] = (_Float16)g3.z; bv[15] = (_Float16)g3.w;

        // ||c_x[m]||^2: each half sums its 16 values, combine across halves.
        float s16 = g0.x*g0.x + g0.y*g0.y + g0.z*g0.z + g0.w*g0.w
                  + g1.x*g1.x + g1.y*g1.y + g1.z*g1.z + g1.w*g1.w
                  + g2.x*g2.x + g2.y*g2.y + g2.z*g2.z + g2.w*g2.w
                  + g3.x*g3.x + g3.y*g3.y + g3.z*g3.z + g3.w*g3.w;
        const float b2v = s16 + __shfl_xor(s16, 16, 32);
        const float cwv = comp_w[m];

        // Partial sum over this lane-half's 64 rows of K^2 for column m.
        float scol = 0.f;
#pragma unroll
        for (int it = 0; it < 8; ++it) {
            v8f c = {};
            c = __builtin_amdgcn_wmma_f32_16x16x32_f16(
                    false, afrag[it], false, bv, (short)0, c, false, false);
#pragma unroll
            for (int v = 0; v < 8; ++v) {
                // C layout: lane = column, row = v + 8*hi within tile.
                const float a2v = s_a2[it * 16 + hi * 8 + v];
                const float d2  = fmaxf(a2v + b2v - 2.0f * c[v], 0.0f);
                // K^2 = exp(-d2/sigma^2) = exp2(-d2 * log2e / sigma^2)
                scol += __builtin_amdgcn_exp2f(-d2 * e2s);
            }
        }

        // Project through comp_w and (c_y/||c_y||)^2 (== c_y^2 / sum(c_y^2)).
        const float* cy = c_y + (size_t)m * DY_;
        float cs = 0.f, cyv[DY_];
#pragma unroll
        for (int d = 0; d < DY_; ++d) { cyv[d] = cy[d]; cs += cyv[d] * cyv[d]; }
        const float w   = cwv * scol;
        const float inv = 1.0f / cs;
        t_loc += w;
#pragma unroll
        for (int d = 0; d < DY_; ++d) p_loc[d] += w * (cyv[d] * cyv[d] * inv);
    }

    // ---- Block reduction (both lane halves hold disjoint row-partials).
    atomicAdd(&s_red[DY_], t_loc);
#pragma unroll
    for (int d = 0; d < DY_; ++d) atomicAdd(&s_red[d], p_loc[d]);
    __syncthreads();

    if (tid < DY_) {
        // out_w includes 1/n and comp_w; 1/n cancels in normalization but is
        // kept for the sum==0 -> EPS edge case fidelity.
        const float tsum = s_red[DY_] * (1.0f / N_);
        const float den  = (tsum == 0.0f) ? 1e-16f : tsum;
        out[(size_t)b * DY_ + tid] = (s_red[tid] * (1.0f / N_)) / den;
    }
}

extern "C" void kernel_launch(void* const* d_in, const int* in_sizes, int n_in,
                              void* d_out, int out_size, void* d_ws, size_t ws_size,
                              hipStream_t stream) {
    const float* inputs = (const float*)d_in[0];
    const float* sigma  = (const float*)d_in[1];
    const float* c_x    = (const float*)d_in[2];
    const float* c_y    = (const float*)d_in[3];
    const float* comp_w = (const float*)d_in[4];
    float* out = (float*)d_out;

    hipLaunchKernelGGL(kqm_fused_kernel, dim3(BS_), dim3(256), 0, stream,
                       inputs, sigma, c_x, c_y, comp_w, out);
}
